// LinearFLH_57123065037500
// MI455X (gfx1250) — compile-verified, matrix operand order checked by reading, not run on hardware
//
#include <hip/hip_runtime.h>

// ---------------------------------------------------------------------------
// Group-quantized linear: out[M,N] = dequant(x)[M,K] @ dequant(w)[N,K]^T
// M = 8192, K = 4096, N = 11008, quant group = 128 along K.
//
// Two-phase plan (avoids re-dequantizing x 86x and w 64x, which would be
// VALU-bound at ~35G lane-ops):
//   pass 1: dequant x -> bf16 x_hat (64 MB), w_int -> bf16 w_hat (86 MB) in ws
//   pass 2: bf16 WMMA GEMM; tile loads are pure copies -> async global->LDS
//           with LDS double buffering; n-major order keeps x_hat hot in L2.
// ---------------------------------------------------------------------------

typedef __attribute__((ext_vector_type(16))) __bf16        v16bf;
typedef __attribute__((ext_vector_type(8)))  float         v8f;
typedef __attribute__((ext_vector_type(4)))  unsigned int  v4u;
typedef __attribute__((ext_vector_type(4)))  int           v4i;
typedef __attribute__((ext_vector_type(4)))  float         v4f;

#define M_TOT 8192
#define N_TOT 11008
#define K_TOT 4096
#define NGRP  32          // K / 128 quant groups
#define BK    128         // K-tile == quant group size
#define LDS_STRIDE 68     // uints per LDS row = 136 bf16 (128 data + 8 pad)

union Frag { v16bf bf; v4u u[2]; };

// ---- float pair -> packed bf16 (RNE) --------------------------------------
__device__ __forceinline__ unsigned int pk2bf(float a, float b) {
#if __has_builtin(__builtin_amdgcn_cvt_pk_bf16_f32)
  typedef __attribute__((ext_vector_type(2))) __bf16 v2bf;
  union { v2bf v; unsigned int u; } r;
  r.v = __builtin_amdgcn_cvt_pk_bf16_f32(a, b);
  return r.u;
#else
  union { float f; unsigned int u; } ua, ub;
  ua.f = a; ub.f = b;
  unsigned int ra = (ua.u + 0x7FFFu + ((ua.u >> 16) & 1u)) >> 16;
  unsigned int rb = (ub.u + 0x7FFFu + ((ub.u >> 16) & 1u)) >> 16;
  return ra | (rb << 16);
#endif
}

// ---- async global->LDS copy (16B per lane), with sync fallback ------------
#if __has_builtin(__builtin_amdgcn_global_load_async_to_lds_b128) && \
    __has_builtin(__builtin_amdgcn_s_wait_asynccnt)
#define HAVE_ASYNC_LDS 1
#else
#define HAVE_ASYNC_LDS 0
#endif

__device__ __forceinline__ void cp16(const void* g, void* l) {
#if HAVE_ASYNC_LDS
  __builtin_amdgcn_global_load_async_to_lds_b128(
      (__attribute__((address_space(1))) v4i*)(void*)g,
      (__attribute__((address_space(3))) v4i*)l, 0, 0);
#else
  *(v4u*)l = *(const v4u*)g;
#endif
}
__device__ __forceinline__ void wait_async_copies() {
#if HAVE_ASYNC_LDS
  __builtin_amdgcn_s_wait_asynccnt(0);
#endif
}

// ---- WMMA operand fragment from LDS tile stored row-major [row][k] --------
// ISA 16-bit A/B layout: lane<16 holds K={k0..k0+7, k0+16..k0+23} of its row,
// lane>=16 holds K={k0+8..k0+15, k0+24..k0+31}.
__device__ __forceinline__ v16bf load_frag(const unsigned int* lds, int row,
                                           int k0, int hi) {
  Frag f;
  const unsigned int* p = lds + row * LDS_STRIDE + (k0 >> 1) + (hi << 2);
  f.u[0] = *(const v4u*)(p);
  f.u[1] = *(const v4u*)(p + 8);
  return f.bf;
}

// ===========================================================================
// Pass 1a: x [M,K] fp32 -> bf16, group-affine dequant. 8 elems / thread.
// ===========================================================================
__global__ __launch_bounds__(256)
void dequant_x_kernel(const float* __restrict__ x,
                      const float* __restrict__ a_scale,
                      const float* __restrict__ a_zero,
                      unsigned short* __restrict__ xh) {
  const size_t gid = (size_t)blockIdx.x * 256 + threadIdx.x;
  const size_t row = gid >> 9;                // / (K/8) = 512
  const int    k   = (int)(gid & 511) << 3;
  const int    kg  = k >> 7;
  const float  sc  = a_scale[row * NGRP + kg];
  const float  bz  = -a_zero[row * NGRP + kg] * sc;
  const v4f* p = (const v4f*)(x + row * K_TOT + k);
  v4f v0 = __builtin_nontemporal_load(p);
  v4f v1 = __builtin_nontemporal_load(p + 1);
  v4u o;
  o.x = pk2bf(fmaf(v0.x, sc, bz), fmaf(v0.y, sc, bz));
  o.y = pk2bf(fmaf(v0.z, sc, bz), fmaf(v0.w, sc, bz));
  o.z = pk2bf(fmaf(v1.x, sc, bz), fmaf(v1.y, sc, bz));
  o.w = pk2bf(fmaf(v1.z, sc, bz), fmaf(v1.w, sc, bz));
  *(v4u*)(xh + row * K_TOT + k) = o;
}

// ===========================================================================
// Pass 1b: w_int [N,K] int32 -> bf16, group-affine dequant. 8 elems / thread.
// ===========================================================================
__global__ __launch_bounds__(256)
void dequant_w_kernel(const int* __restrict__ w_int,
                      const float* __restrict__ w_scale,
                      const float* __restrict__ w_zero,
                      unsigned short* __restrict__ wh) {
  const size_t gid = (size_t)blockIdx.x * 256 + threadIdx.x;
  const size_t row = gid >> 9;
  const int    k   = (int)(gid & 511) << 3;
  const int    kg  = k >> 7;
  const float  sc  = w_scale[row * NGRP + kg];
  const float  bz  = -w_zero[row * NGRP + kg] * sc;
  const v4i* p = (const v4i*)(w_int + row * K_TOT + k);
  v4i q0 = __builtin_nontemporal_load(p);
  v4i q1 = __builtin_nontemporal_load(p + 1);
  v4u o;
  o.x = pk2bf(fmaf((float)q0.x, sc, bz), fmaf((float)q0.y, sc, bz));
  o.y = pk2bf(fmaf((float)q0.z, sc, bz), fmaf((float)q0.w, sc, bz));
  o.z = pk2bf(fmaf((float)q1.x, sc, bz), fmaf((float)q1.y, sc, bz));
  o.w = pk2bf(fmaf((float)q1.z, sc, bz), fmaf((float)q1.w, sc, bz));
  *(v4u*)(wh + row * K_TOT + k) = o;
}

// ===========================================================================
// Pass 2: bf16 WMMA GEMM with async global->LDS double buffering.
// 128x128 block tile, 8 waves each computing 32(M) x 64(N) via 2x4 WMMA accs.
// ===========================================================================
__global__ __launch_bounds__(256)
void gemm_bf16_kernel(const unsigned short* __restrict__ xh,
                      const unsigned short* __restrict__ wh,
                      float* __restrict__ out) {
  __shared__ unsigned int ldsA[2][128 * LDS_STRIDE];
  __shared__ unsigned int ldsB[2][128 * LDS_STRIDE];

  constexpr int MT = M_TOT / 128;   // 64
  constexpr int KT = K_TOT / BK;    // 32

  const int tid = threadIdx.x;

  // n-major ordering: the entire 64 MB x_hat stays resident in L2; each
  // group of 64 consecutive blocks shares one 1 MB w_hat tile column.
  const int mt = blockIdx.x % MT;
  const int nt = blockIdx.x / MT;

  // loader coords: 2 threads per row, 64 contiguous bf16 (8 x 16B) each
  const int r = tid >> 1;
  const int h = tid & 1;
  const size_t gm = (size_t)mt * 128 + r;
  const size_t gn = (size_t)nt * 128 + r;
  const unsigned short* gArow = xh + gm * K_TOT + h * 64;
  const unsigned short* gBrow = wh + gn * K_TOT + h * 64;
  const int ldsOff = r * LDS_STRIDE + h * 32;   // uints

  // wave coords
  const int wave = tid >> 5;
  const int wm   = wave >> 1;          // 0..3 -> M offset wm*32
  const int wn   = wave & 1;           // 0..1 -> N offset wn*64
  const int lane = tid & 31;
  const int lo   = lane & 15;
  const int hi   = lane >> 4;

  const v8f vzero = {0.f, 0.f, 0.f, 0.f, 0.f, 0.f, 0.f, 0.f};
  v8f acc[2][4];
#pragma unroll
  for (int i = 0; i < 2; ++i)
#pragma unroll
    for (int j = 0; j < 4; ++j) acc[i][j] = vzero;

  auto issue = [&](int buf, int kg) {
    const unsigned short* ga = gArow + (size_t)kg * BK;
    const unsigned short* gb = gBrow + (size_t)kg * BK;
    unsigned int* la = &ldsA[buf][ldsOff];
    unsigned int* lb = &ldsB[buf][ldsOff];
#pragma unroll
    for (int i = 0; i < 8; ++i) {
      cp16(ga + 8 * i, la + 4 * i);
      cp16(gb + 8 * i, lb + 4 * i);
    }
  };

  issue(0, 0);
  wait_async_copies();
  __syncthreads();

  for (int kg = 0; kg < KT; ++kg) {
    const int cb = kg & 1;
    if (kg + 1 < KT) issue(cb ^ 1, kg + 1);   // prefetch next tile into LDS

    const unsigned int* tA = &ldsA[cb][0];
    const unsigned int* tB = &ldsB[cb][0];
#pragma unroll
    for (int ks = 0; ks < 4; ++ks) {          // K = 128 -> 4 steps of 32
      const int k0 = ks * 32;
      v16bf fa0 = load_frag(tA, wm * 32 +  0 + lo, k0, hi);
      v16bf fa1 = load_frag(tA, wm * 32 + 16 + lo, k0, hi);
      v16bf fb0 = load_frag(tB, wn * 64 +  0 + lo, k0, hi);
      v16bf fb1 = load_frag(tB, wn * 64 + 16 + lo, k0, hi);
      v16bf fb2 = load_frag(tB, wn * 64 + 32 + lo, k0, hi);
      v16bf fb3 = load_frag(tB, wn * 64 + 48 + lo, k0, hi);
      acc[0][0] = __builtin_amdgcn_wmma_f32_16x16x32_bf16(false, fa0, false, fb0, (short)0, acc[0][0], false, false);
      acc[0][1] = __builtin_amdgcn_wmma_f32_16x16x32_bf16(false, fa0, false, fb1, (short)0, acc[0][1], false, false);
      acc[0][2] = __builtin_amdgcn_wmma_f32_16x16x32_bf16(false, fa0, false, fb2, (short)0, acc[0][2], false, false);
      acc[0][3] = __builtin_amdgcn_wmma_f32_16x16x32_bf16(false, fa0, false, fb3, (short)0, acc[0][3], false, false);
      acc[1][0] = __builtin_amdgcn_wmma_f32_16x16x32_bf16(false, fa1, false, fb0, (short)0, acc[1][0], false, false);
      acc[1][1] = __builtin_amdgcn_wmma_f32_16x16x32_bf16(false, fa1, false, fb1, (short)0, acc[1][1], false, false);
      acc[1][2] = __builtin_amdgcn_wmma_f32_16x16x32_bf16(false, fa1, false, fb2, (short)0, acc[1][2], false, false);
      acc[1][3] = __builtin_amdgcn_wmma_f32_16x16x32_bf16(false, fa1, false, fb3, (short)0, acc[1][3], false, false);
    }
    wait_async_copies();   // my copies for kg+1 done (no-op on fallback)
    __syncthreads();       // everyone done reading cb / filling cb^1
  }

  // epilogue: ISA 16x16 f32 C/D layout (VGPR e -> M = e + 8*hi, lane -> N);
  // nontemporal stores keep the 360 MB output from thrashing x_hat in L2.
#pragma unroll
  for (int i = 0; i < 2; ++i) {
#pragma unroll
    for (int j = 0; j < 4; ++j) {
      const size_t orow = (size_t)mt * 128 + wm * 32 + i * 16 + hi * 8;
      const size_t ocol = (size_t)nt * 128 + wn * 64 + j * 16 + lo;
      float* op = out + orow * N_TOT + ocol;
#pragma unroll
      for (int e = 0; e < 8; ++e)
        __builtin_nontemporal_store(acc[i][j][e], op + (size_t)e * N_TOT);
    }
  }
}

// ===========================================================================
// Fallback: fused dequant+GEMM (round-1 kernel) if workspace is too small.
// ===========================================================================
__global__ __launch_bounds__(256)
void fused_gemm_kernel(const float* __restrict__ x,
                       const float* __restrict__ a_scale,
                       const float* __restrict__ a_zero,
                       const int*   __restrict__ w_int,
                       const float* __restrict__ w_scale,
                       const float* __restrict__ w_zero,
                       float*       __restrict__ out) {
  __shared__ unsigned int ldsA[128 * LDS_STRIDE];
  __shared__ unsigned int ldsB[128 * LDS_STRIDE];

  constexpr int NT = N_TOT / 128;
  constexpr int KT = K_TOT / BK;
  const int tid = threadIdx.x;

  const int GMG = 16;
  int bid = blockIdx.x;
  int per_group = GMG * NT;
  int grp = bid / per_group;
  int rem = bid - grp * per_group;
  const int mt = grp * GMG + (rem % GMG);
  const int nt = rem / GMG;

  const int r    = tid >> 1;
  const int col0 = (tid & 1) << 6;
  const size_t gm = (size_t)mt * 128 + r;
  const size_t gn = (size_t)nt * 128 + r;
  const float* xrow = x     + gm * K_TOT + col0;
  const int*   wrow = w_int + gn * K_TOT + col0;
  unsigned int* la = &ldsA[r * LDS_STRIDE + (col0 >> 1)];
  unsigned int* lb = &ldsB[r * LDS_STRIDE + (col0 >> 1)];

  const int wave = tid >> 5;
  const int wm   = wave >> 1;
  const int wn   = wave & 1;
  const int lane = tid & 31;
  const int lo   = lane & 15;
  const int hi   = lane >> 4;

  const v8f vzero = {0.f, 0.f, 0.f, 0.f, 0.f, 0.f, 0.f, 0.f};
  v8f acc[2][4];
#pragma unroll
  for (int i = 0; i < 2; ++i)
#pragma unroll
    for (int j = 0; j < 4; ++j) acc[i][j] = vzero;

  for (int kg = 0; kg < KT; ++kg) {
    const float as = a_scale[gm * NGRP + kg];
    const float az = a_zero [gm * NGRP + kg];
    const float ws = w_scale[gn * NGRP + kg];
    const float wz = w_zero [gn * NGRP + kg];
    const v4f* xp = (const v4f*)(xrow + kg * BK);
    const v4i* wp = (const v4i*)(wrow + kg * BK);
#pragma unroll
    for (int it = 0; it < 8; ++it) {
      v4f v0 = xp[2 * it], v1 = xp[2 * it + 1];
      v4i q0 = wp[2 * it], q1 = wp[2 * it + 1];
      v4u pa, pb;
      pa.x = pk2bf((v0.x - az) * as, (v0.y - az) * as);
      pa.y = pk2bf((v0.z - az) * as, (v0.w - az) * as);
      pa.z = pk2bf((v1.x - az) * as, (v1.y - az) * as);
      pa.w = pk2bf((v1.z - az) * as, (v1.w - az) * as);
      *(v4u*)(la + 4 * it) = pa;
      pb.x = pk2bf(((float)q0.x - wz) * ws, ((float)q0.y - wz) * ws);
      pb.y = pk2bf(((float)q0.z - wz) * ws, ((float)q0.w - wz) * ws);
      pb.z = pk2bf(((float)q1.x - wz) * ws, ((float)q1.y - wz) * ws);
      pb.w = pk2bf(((float)q1.z - wz) * ws, ((float)q1.w - wz) * ws);
      *(v4u*)(lb + 4 * it) = pb;
    }
    __syncthreads();
#pragma unroll
    for (int ks = 0; ks < 4; ++ks) {
      const int k0 = ks * 32;
      v16bf fa0 = load_frag(ldsA, wm * 32 +  0 + lo, k0, hi);
      v16bf fa1 = load_frag(ldsA, wm * 32 + 16 + lo, k0, hi);
      v16bf fb0 = load_frag(ldsB, wn * 64 +  0 + lo, k0, hi);
      v16bf fb1 = load_frag(ldsB, wn * 64 + 16 + lo, k0, hi);
      v16bf fb2 = load_frag(ldsB, wn * 64 + 32 + lo, k0, hi);
      v16bf fb3 = load_frag(ldsB, wn * 64 + 48 + lo, k0, hi);
      acc[0][0] = __builtin_amdgcn_wmma_f32_16x16x32_bf16(false, fa0, false, fb0, (short)0, acc[0][0], false, false);
      acc[0][1] = __builtin_amdgcn_wmma_f32_16x16x32_bf16(false, fa0, false, fb1, (short)0, acc[0][1], false, false);
      acc[0][2] = __builtin_amdgcn_wmma_f32_16x16x32_bf16(false, fa0, false, fb2, (short)0, acc[0][2], false, false);
      acc[0][3] = __builtin_amdgcn_wmma_f32_16x16x32_bf16(false, fa0, false, fb3, (short)0, acc[0][3], false, false);
      acc[1][0] = __builtin_amdgcn_wmma_f32_16x16x32_bf16(false, fa1, false, fb0, (short)0, acc[1][0], false, false);
      acc[1][1] = __builtin_amdgcn_wmma_f32_16x16x32_bf16(false, fa1, false, fb1, (short)0, acc[1][1], false, false);
      acc[1][2] = __builtin_amdgcn_wmma_f32_16x16x32_bf16(false, fa1, false, fb2, (short)0, acc[1][2], false, false);
      acc[1][3] = __builtin_amdgcn_wmma_f32_16x16x32_bf16(false, fa1, false, fb3, (short)0, acc[1][3], false, false);
    }
    __syncthreads();
  }
#pragma unroll
  for (int i = 0; i < 2; ++i) {
#pragma unroll
    for (int j = 0; j < 4; ++j) {
      const size_t orow = (size_t)mt * 128 + wm * 32 + i * 16 + hi * 8;
      const size_t ocol = (size_t)nt * 128 + wn * 64 + j * 16 + lo;
      float* op = out + orow * N_TOT + ocol;
#pragma unroll
      for (int e = 0; e < 8; ++e)
        op[(size_t)e * N_TOT] = acc[i][j][e];
    }
  }
}

// ===========================================================================
extern "C" void kernel_launch(void* const* d_in, const int* in_sizes, int n_in,
                              void* d_out, int out_size, void* d_ws, size_t ws_size,
                              hipStream_t stream) {
  const float* x       = (const float*)d_in[0];
  const float* a_scale = (const float*)d_in[1];
  const float* a_zero  = (const float*)d_in[2];
  const int*   w_int   = (const int*)  d_in[3];
  const float* w_scale = (const float*)d_in[4];
  const float* w_zero  = (const float*)d_in[5];
  float* out = (float*)d_out;

  const size_t xh_bytes = (size_t)M_TOT * K_TOT * 2;   // 64 MB
  const size_t wh_bytes = (size_t)N_TOT * K_TOT * 2;   // 86 MB
  const int mtiles = M_TOT / 128;   // 64
  const int ntiles = N_TOT / 128;   // 86

  if (ws_size >= xh_bytes + wh_bytes) {
    unsigned short* xh = (unsigned short*)d_ws;
    unsigned short* wh = (unsigned short*)((char*)d_ws + xh_bytes);
    dequant_x_kernel<<<dim3((unsigned)((M_TOT * (size_t)K_TOT / 8) / 256)), dim3(256), 0, stream>>>(
        x, a_scale, a_zero, xh);
    dequant_w_kernel<<<dim3((unsigned)((N_TOT * (size_t)K_TOT / 8) / 256)), dim3(256), 0, stream>>>(
        w_int, w_scale, w_zero, wh);
    gemm_bf16_kernel<<<dim3(mtiles * ntiles), dim3(256), 0, stream>>>(xh, wh, out);
  } else {
    fused_gemm_kernel<<<dim3(mtiles * ntiles), dim3(256), 0, stream>>>(
        x, a_scale, a_zero, w_int, w_scale, w_zero, out);
  }
}
